// Block_46196668236003
// MI455X (gfx1250) — matrix operand order repaired and checked by reading, not compiled, not run on hardware
//
#include <hip/hip_runtime.h>
#include <math.h>

// ---------------- problem constants (B=2, T=2048, C=1024) ----------------
constexpr int B_  = 2;
constexpr int T_  = 2048;
constexpr int C_  = 1024;
constexpr int H_  = 16;
constexpr int HS_ = 64;
constexpr int M_  = B_ * T_;        // 4096 tokens
constexpr int FF_ = 4 * C_;         // 4096
constexpr int DW_ = 64, DA_ = 64, DV_ = 32, DG_ = 160;

// ---------------- WMMA vector types ----------------
typedef __attribute__((ext_vector_type(16))) __bf16 v16bf;
typedef __attribute__((ext_vector_type(8)))  __bf16 v8bf;
typedef __attribute__((ext_vector_type(8)))  float  v8f;

static __device__ __forceinline__ float sigmoidf_(float x) {
  return 1.0f / (1.0f + expf(-x));
}

static __device__ __forceinline__ float wave_sum32(float v) {
  v += __shfl_xor(v, 1, 32);
  v += __shfl_xor(v, 2, 32);
  v += __shfl_xor(v, 4, 32);
  v += __shfl_xor(v, 8, 32);
  v += __shfl_xor(v, 16, 32);
  return v;
}

// =====================================================================
// Tiled bf16 WMMA GEMM:  Cc[M,N] (f32) = A[M,K] (bf16, row-major)
//                                       x Bw[N,K] (bf16, row-major, i.e. B^T)
// Block = 128 threads = 4 waves; each wave computes a 32x64 tile:
//   2 M-subtiles x 4 N-subtiles = 8 independent WMMA accumulator chains,
//   each B fragment reused by 2 WMMAs.
// OOB N-columns load a clamped dummy row (their product lands only in the
// store-guarded lane), so the inner K loop is completely branch-free.
// Fragment packing: contiguous k order (element s of lane half h holds k = 16h + s) for A and B.
// Requires: M % 128 == 0, K % 32 == 0.
// =====================================================================
__global__ __launch_bounds__(128)
void k_gemm_bf16(const __bf16* __restrict__ A, const __bf16* __restrict__ Bw,
                 float* __restrict__ Cc, int M, int N, int K)
{
  const int lane = threadIdx.x & 31;
  const int wave = threadIdx.x >> 5;
  const int mBase = blockIdx.y * 128 + wave * 32;   // two 16-row subtiles
  const int n0    = blockIdx.x * 64;
  const int half  = lane >> 4;      // 0 => lanes 0-15, 1 => lanes 16-31
  const int l15   = lane & 15;

  const __bf16* Ab0 = A + (size_t)(mBase + l15) * K + half * 16;
  const __bf16* Ab1 = Ab0 + (size_t)16 * K;

  // B pointers per N-subtile, clamped to a valid row for OOB columns.
  const __bf16* Bp[4];
#pragma unroll
  for (int nt = 0; nt < 4; ++nt) {
    const int ncol = n0 + nt * 16 + l15;
    const int nc   = (ncol < N) ? ncol : (N - 1);
    Bp[nt] = Bw + (size_t)nc * K + half * 16;
  }

  v8f acc[8];
#pragma unroll
  for (int q = 0; q < 8; ++q)
#pragma unroll
    for (int j = 0; j < 8; ++j) acc[q][j] = 0.0f;

  for (int k0 = 0; k0 < K; k0 += 32) {
    union { v16bf v; v8bf h[2]; } a0, a1;
    a0.h[0] = *(const v8bf*)(Ab0 + k0);        // K: k0 + half*16 .. +7
    a0.h[1] = *(const v8bf*)(Ab0 + k0 + 8);    // K: k0+8+half*16 .. +7
    a1.h[0] = *(const v8bf*)(Ab1 + k0);
    a1.h[1] = *(const v8bf*)(Ab1 + k0 + 8);
#pragma unroll
    for (int nt = 0; nt < 4; ++nt) {
      const v16bf bf = *(const v16bf*)(Bp[nt] + k0);
      acc[nt] = __builtin_amdgcn_wmma_f32_16x16x32_bf16(
          false, a0.v, false, bf, (short)0, acc[nt], false, false);
      acc[4 + nt] = __builtin_amdgcn_wmma_f32_16x16x32_bf16(
          false, a1.v, false, bf, (short)0, acc[4 + nt], false, false);
    }
  }

#pragma unroll
  for (int mt = 0; mt < 2; ++mt) {
    const int rbase = mBase + mt * 16 + half * 8;
#pragma unroll
    for (int nt = 0; nt < 4; ++nt) {
      const int ncol = n0 + nt * 16 + l15;
      if (ncol < N) {
#pragma unroll
        for (int j = 0; j < 8; ++j)
          Cc[(size_t)(rbase + j) * N + ncol] = acc[mt * 4 + nt][j];
      }
    }
  }
}

// ---------------- LayerNorm over rows of length Crow ----------------
__global__ __launch_bounds__(256)
void k_layernorm(const float* __restrict__ x, const float* __restrict__ w,
                 const float* __restrict__ b, float* __restrict__ out, int Crow)
{
  const int row = blockIdx.x;
  const float* xr = x + (size_t)row * Crow;
  float s = 0.f, s2 = 0.f;
  for (int c = threadIdx.x; c < Crow; c += blockDim.x) {
    float v = xr[c]; s += v; s2 += v * v;
  }
  s = wave_sum32(s); s2 = wave_sum32(s2);
  __shared__ float w1s[8], w2s[8];
  if ((threadIdx.x & 31) == 0) { w1s[threadIdx.x >> 5] = s; w2s[threadIdx.x >> 5] = s2; }
  __syncthreads();
  float tot = 0.f, tot2 = 0.f;
#pragma unroll
  for (int i = 0; i < 8; ++i) { tot += w1s[i]; tot2 += w2s[i]; }
  const float mu  = tot / Crow;
  const float var = tot2 / Crow - mu * mu;
  const float inv = rsqrtf(var + 1e-5f);
  float* o = out + (size_t)row * Crow;
  for (int c = threadIdx.x; c < Crow; c += blockDim.x)
    o[c] = (xr[c] - mu) * inv * w[c] + b[c];
}

// ---------------- token-shift mixes (6-way), bf16 outputs ----------------
__global__ __launch_bounds__(256)
void k_mix6(const float* __restrict__ xn,
            const float* __restrict__ mr, const float* __restrict__ mw,
            const float* __restrict__ mk, const float* __restrict__ mv,
            const float* __restrict__ ma, const float* __restrict__ mg,
            __bf16* __restrict__ Xr, __bf16* __restrict__ Xw, __bf16* __restrict__ Xk,
            __bf16* __restrict__ Xv, __bf16* __restrict__ Xa, __bf16* __restrict__ Xg)
{
  size_t idx = (size_t)blockIdx.x * blockDim.x + threadIdx.x;
  if (idx >= (size_t)M_ * C_) return;
  const int c = (int)(idx % C_);
  const int t = (int)((idx / C_) % T_);
  const float cur  = xn[idx];
  const float prev = (t == 0) ? 0.f : xn[idx - C_];
  const float d = prev - cur;
  Xr[idx] = (__bf16)(cur + d * mr[c]);
  Xw[idx] = (__bf16)(cur + d * mw[c]);
  Xk[idx] = (__bf16)(cur + d * mk[c]);
  Xv[idx] = (__bf16)(cur + d * mv[c]);
  Xa[idx] = (__bf16)(cur + d * ma[c]);
  Xg[idx] = (__bf16)(cur + d * mg[c]);
}

__global__ __launch_bounds__(256)
void k_mix1(const float* __restrict__ xn, const float* __restrict__ mk,
            __bf16* __restrict__ out)
{
  size_t idx = (size_t)blockIdx.x * blockDim.x + threadIdx.x;
  if (idx >= (size_t)M_ * C_) return;
  const int c = (int)(idx % C_);
  const int t = (int)((idx / C_) % T_);
  const float cur  = xn[idx];
  const float prev = (t == 0) ? 0.f : xn[idx - C_];
  out[idx] = (__bf16)(cur + (prev - cur) * mk[c]);
}

// ---------------- conversions / activations to bf16 ----------------
__global__ __launch_bounds__(256)
void k_f2b(const float* __restrict__ in, __bf16* __restrict__ out, size_t n)
{
  size_t i = (size_t)blockIdx.x * blockDim.x + threadIdx.x;
  if (i < n) out[i] = (__bf16)in[i];
}
__global__ __launch_bounds__(256)
void k_tanh_f2b(const float* __restrict__ in, __bf16* __restrict__ out, size_t n)
{
  size_t i = (size_t)blockIdx.x * blockDim.x + threadIdx.x;
  if (i < n) out[i] = (__bf16)tanhf(in[i]);
}
__global__ __launch_bounds__(256)
void k_sigm_f2b(const float* __restrict__ in, __bf16* __restrict__ out, size_t n)
{
  size_t i = (size_t)blockIdx.x * blockDim.x + threadIdx.x;
  if (i < n) out[i] = (__bf16)sigmoidf_(in[i]);
}
__global__ __launch_bounds__(256)
void k_relu2_f2b(const float* __restrict__ in, __bf16* __restrict__ out, size_t n)
{
  size_t i = (size_t)blockIdx.x * blockDim.x + threadIdx.x;
  if (i < n) { float t = fmaxf(in[i], 0.f); out[i] = (__bf16)(t * t); }
}
// transpose convert: src [R, Cc] f32 -> dst [Cc, R] bf16
__global__ __launch_bounds__(256)
void k_tr_f2b(const float* __restrict__ src, __bf16* __restrict__ dst, int R, int Cc)
{
  size_t i = (size_t)blockIdx.x * blockDim.x + threadIdx.x;
  if (i >= (size_t)R * Cc) return;
  const int r = (int)(i / Cc), c = (int)(i % Cc);
  dst[(size_t)c * R + r] = (__bf16)src[i];
}
__global__ __launch_bounds__(256)
void k_add(const float* __restrict__ a, const float* __restrict__ b,
           float* __restrict__ out, size_t n)
{
  size_t i = (size_t)blockIdx.x * blockDim.x + threadIdx.x;
  if (i < n) out[i] = a[i] + b[i];
}
__global__ __launch_bounds__(256)
void k_copy(const float* __restrict__ a, float* __restrict__ out, size_t n)
{
  size_t i = (size_t)blockIdx.x * blockDim.x + threadIdx.x;
  if (i < n) out[i] = a[i];
}

// ---------------- scan input prep: one wave per (token, head) ----------------
__global__ __launch_bounds__(256)
void k_prep_scan(const float* __restrict__ k, const float* __restrict__ vraw,
                 const float* __restrict__ vl2, const float* __restrict__ wl2,
                 const float* __restrict__ apre, const float* __restrict__ vfirst,
                 const float* __restrict__ kkv, const float* __restrict__ kav,
                 const float* __restrict__ a0v, const float* __restrict__ v0v,
                 const float* __restrict__ w0v,
                 float* __restrict__ decay, float* __restrict__ kout,
                 float* __restrict__ vout, float* __restrict__ aaout,
                 float* __restrict__ bbout, int total_heads)
{
  const int gwave = (int)((blockIdx.x * (size_t)blockDim.x + threadIdx.x) >> 5);
  if (gwave >= total_heads) return;
  const int lane = threadIdx.x & 31;
  const int h = gwave % H_;
  const int tok = gwave / H_;
  const size_t base = (size_t)tok * C_ + h * HS_;

  float nrm2 = 0.f, kkq[2], aq[2], knq[2], vq[2], dq[2];
#pragma unroll
  for (int q = 0; q < 2; ++q) {
    const int i = lane * 2 + q;
    const size_t idx = base + i;
    const int c = h * HS_ + i;
    const float kx = k[idx];
    const float a  = sigmoidf_(a0v[c] + apre[idx]);
    const float sv = sigmoidf_(v0v[c] + vl2[idx]);
    const float vr = vraw[idx];
    const float vv = vr + (vfirst[idx] - vr) * sv;
    const float u  = w0v[c] + wl2[idx];
    const float w  = -log1pf(expf(-u)) - 0.5f;      // -softplus(-u) - 0.5
    const float kkr = kx * kkv[c];
    nrm2 += kkr * kkr;
    kkq[q] = kkr; aq[q] = a;
    knq[q] = kx * (1.f + (a - 1.f) * kav[c]);
    vq[q]  = vv;
    dq[q]  = expf(w);
  }
  nrm2 = wave_sum32(nrm2);
  const float inv = 1.f / fmaxf(sqrtf(nrm2), 1e-12f);
#pragma unroll
  for (int q = 0; q < 2; ++q) {
    const size_t idx = base + lane * 2 + q;
    const float kk = kkq[q] * inv;
    decay[idx] = dq[q];
    kout[idx]  = knq[q];
    vout[idx]  = vq[q];
    aaout[idx] = -kk;
    bbout[idx] = kk * aq[q];
  }
}

// ---------------- WKV7 recurrence: one block per (b,h) ----------------
// 256 threads; thread owns 16 state elems: row i = tid>>2, cols j0..j0+15.
__global__ __launch_bounds__(256)
void k_wkv7_scan(const float* __restrict__ S0, const float* __restrict__ decay,
                 const float* __restrict__ r, const float* __restrict__ k,
                 const float* __restrict__ v, const float* __restrict__ aa,
                 const float* __restrict__ bb, float* __restrict__ y)
{
  const int bh = blockIdx.x;
  const int b = bh / H_, h = bh % H_;
  const int tid = threadIdx.x;
  const int i  = tid >> 2;
  const int jq = tid & 3;
  const int j0 = jq * 16;

  float S[16];
  const float* s0 = S0 + (size_t)bh * HS_ * HS_;
#pragma unroll
  for (int m = 0; m < 16; ++m) S[m] = s0[(size_t)i * HS_ + j0 + m];

  __shared__ float sw[64], sr[64], sk[64], sv[64], sa[64], sb[64];

  for (int t = 0; t < T_; ++t) {
    const size_t base = ((size_t)(b * T_ + t) * H_ + h) * HS_;
    {
      const int ix = tid & 63;
      if (tid < 64)        sw[ix] = decay[base + ix];
      else if (tid < 128)  sr[ix] = r[base + ix];
      else if (tid < 192)  sk[ix] = k[base + ix];
      else                 sv[ix] = v[base + ix];
      if (tid < 64)        sa[ix] = aa[base + ix];
      else if (tid < 128)  sb[ix] = bb[base + ix];
    }
    __syncthreads();

    float sap = 0.f;
#pragma unroll
    for (int m = 0; m < 16; ++m) sap += S[m] * sa[j0 + m];
    sap += __shfl_xor(sap, 1, 32);
    sap += __shfl_xor(sap, 2, 32);

    const float vi = sv[i];
    float yp = 0.f;
#pragma unroll
    for (int m = 0; m < 16; ++m) {
      float s = S[m] * sw[j0 + m] + sap * sb[j0 + m] + vi * sk[j0 + m];
      S[m] = s;
      yp += s * sr[j0 + m];
    }
    yp += __shfl_xor(yp, 1, 32);
    yp += __shfl_xor(yp, 2, 32);
    if (jq == 0) y[base + i] = yp;
    __syncthreads();
  }
}

// ---------------- fused GroupNorm + r.k.r_k bonus + gating ----------------
__global__ __launch_bounds__(256)
void k_gn_gate(const float* __restrict__ y, const float* __restrict__ r,
               const float* __restrict__ k, const float* __restrict__ v,
               const float* __restrict__ r_k, const float* __restrict__ lnxw,
               const float* __restrict__ lnxb, const float* __restrict__ g,
               __bf16* __restrict__ yg, int total_heads)
{
  const int gwave = (int)((blockIdx.x * (size_t)blockDim.x + threadIdx.x) >> 5);
  if (gwave >= total_heads) return;
  const int lane = threadIdx.x & 31;
  const int h = gwave % H_;
  const int tok = gwave / H_;
  const size_t base = (size_t)tok * C_ + h * HS_;

  float yv[2], s = 0.f, s2 = 0.f, dp = 0.f;
#pragma unroll
  for (int q = 0; q < 2; ++q) {
    const int i = lane * 2 + q;
    const float t = y[base + i];
    yv[q] = t; s += t; s2 += t * t;
    dp += r[base + i] * k[base + i] * r_k[h * HS_ + i];
  }
  s = wave_sum32(s); s2 = wave_sum32(s2); dp = wave_sum32(dp);
  const float mu  = s / HS_;
  const float var = s2 / HS_ - mu * mu;
  const float inv = rsqrtf(var + 64e-5f);
#pragma unroll
  for (int q = 0; q < 2; ++q) {
    const int i = lane * 2 + q;
    const int c = h * HS_ + i;
    float yy = (yv[q] - mu) * inv * lnxw[c] + lnxb[c];
    yy += dp * v[base + i];
    yg[base + i] = (__bf16)(yy * g[base + i]);
  }
}

// =====================================================================
// host-side orchestration
// =====================================================================
static inline dim3 ew_grid(size_t n) { return dim3((unsigned)((n + 255) / 256)); }

extern "C" void kernel_launch(void* const* d_in, const int* in_sizes, int n_in,
                              void* d_out, int out_size, void* d_ws, size_t ws_size,
                              hipStream_t stream)
{
  (void)in_sizes; (void)n_in; (void)out_size; (void)ws_size;
  const float* x      = (const float*)d_in[0];
  const float* vfirst = (const float*)d_in[1];
  const float* s0     = (const float*)d_in[2];
  const float* ln1w   = (const float*)d_in[3];
  const float* ln1b   = (const float*)d_in[4];
  const float* ln2w   = (const float*)d_in[5];
  const float* ln2b   = (const float*)d_in[6];
  const float* x_r    = (const float*)d_in[7];
  const float* x_w    = (const float*)d_in[8];
  const float* x_k    = (const float*)d_in[9];
  const float* x_v    = (const float*)d_in[10];
  const float* x_a    = (const float*)d_in[11];
  const float* x_g    = (const float*)d_in[12];
  const float* w0     = (const float*)d_in[13];
  const float* w1     = (const float*)d_in[14];
  const float* w2     = (const float*)d_in[15];
  const float* a0     = (const float*)d_in[16];
  const float* a1     = (const float*)d_in[17];
  const float* a2     = (const float*)d_in[18];
  const float* v0     = (const float*)d_in[19];
  const float* v1     = (const float*)d_in[20];
  const float* v2     = (const float*)d_in[21];
  const float* g1     = (const float*)d_in[22];
  const float* g2     = (const float*)d_in[23];
  const float* k_k    = (const float*)d_in[24];
  const float* k_a    = (const float*)d_in[25];
  const float* r_k    = (const float*)d_in[26];
  const float* W_r    = (const float*)d_in[27];
  const float* W_k    = (const float*)d_in[28];
  const float* W_v    = (const float*)d_in[29];
  const float* W_o    = (const float*)d_in[30];
  const float* lnxw   = (const float*)d_in[31];
  const float* lnxb   = (const float*)d_in[32];
  const float* mixk   = (const float*)d_in[33];
  const float* Wkey   = (const float*)d_in[34];
  const float* Wval   = (const float*)d_in[35];

  char* ws = (char*)d_ws;
  size_t off = 0;
  auto alloc = [&](size_t bytes) -> void* {
    void* p = ws + off;
    off += bytes;
    off = (off + 255) & ~(size_t)255;
    return p;
  };
  const size_t MC = (size_t)M_ * C_;

  float*  xn    = (float*) alloc(MC * 4);
  __bf16* Xr    = (__bf16*)alloc(MC * 2);
  __bf16* Xw    = (__bf16*)alloc(MC * 2);
  __bf16* Xk    = (__bf16*)alloc(MC * 2);
  __bf16* Xv    = (__bf16*)alloc(MC * 2);
  __bf16* Xa    = (__bf16*)alloc(MC * 2);
  __bf16* Xg    = (__bf16*)alloc(MC * 2);
  __bf16* bWr   = (__bf16*)alloc((size_t)C_ * C_ * 2);
  __bf16* bWk   = (__bf16*)alloc((size_t)C_ * C_ * 2);
  __bf16* bWv   = (__bf16*)alloc((size_t)C_ * C_ * 2);
  __bf16* bWo   = (__bf16*)alloc((size_t)C_ * C_ * 2);
  __bf16* bW1   = (__bf16*)alloc((size_t)C_ * DW_ * 2);
  __bf16* bW2   = (__bf16*)alloc((size_t)C_ * DW_ * 2);
  __bf16* bA1   = (__bf16*)alloc((size_t)C_ * DA_ * 2);
  __bf16* bA2   = (__bf16*)alloc((size_t)C_ * DA_ * 2);
  __bf16* bV1   = (__bf16*)alloc((size_t)C_ * DV_ * 2);
  __bf16* bV2   = (__bf16*)alloc((size_t)C_ * DV_ * 2);
  __bf16* bG1   = (__bf16*)alloc((size_t)C_ * DG_ * 2);
  __bf16* bG2   = (__bf16*)alloc((size_t)C_ * DG_ * 2);
  __bf16* bWkey = (__bf16*)alloc((size_t)FF_ * C_ * 2);
  __bf16* bWval = (__bf16*)alloc((size_t)C_ * FF_ * 2);
  float*  rbuf  = (float*) alloc(MC * 4);
  float*  kbuf  = (float*) alloc(MC * 4);
  float*  vraw  = (float*) alloc(MC * 4);
  float*  wl1   = (float*) alloc((size_t)M_ * DW_ * 4);
  __bf16* wl1b  = (__bf16*)alloc((size_t)M_ * DW_ * 2);
  float*  wl2   = (float*) alloc(MC * 4);
  float*  al1   = (float*) alloc((size_t)M_ * DA_ * 4);
  __bf16* al1b  = (__bf16*)alloc((size_t)M_ * DA_ * 2);
  float*  al2   = (float*) alloc(MC * 4);
  float*  vl1   = (float*) alloc((size_t)M_ * DV_ * 4);
  __bf16* vl1b  = (__bf16*)alloc((size_t)M_ * DV_ * 2);
  float*  vl2   = (float*) alloc(MC * 4);
  float*  gl1   = (float*) alloc((size_t)M_ * DG_ * 4);
  __bf16* gl1b  = (__bf16*)alloc((size_t)M_ * DG_ * 2);
  float*  gbuf  = (float*) alloc(MC * 4);
  float*  decay = (float*) alloc(MC * 4);
  float*  knew  = (float*) alloc(MC * 4);
  float*  vfin  = (float*) alloc(MC * 4);
  float*  aab   = (float*) alloc(MC * 4);
  float*  bbb   = (float*) alloc(MC * 4);
  float*  ybuf  = (float*) alloc(MC * 4);
  __bf16* ygb   = (__bf16*)alloc(MC * 2);
  float*  att   = (float*) alloc(MC * 4);
  float*  x1    = (float*) alloc(MC * 4);
  float*  xn2   = (float*) alloc(MC * 4);
  __bf16* kfb   = (__bf16*)alloc(MC * 2);
  float*  ffn1  = (float*) alloc((size_t)M_ * FF_ * 4);
  __bf16* ffn1b = (__bf16*)alloc((size_t)M_ * FF_ * 2);
  float*  ffn2  = (float*) alloc(MC * 4);

  // ---- weight conversion to bf16 (LoRA mats transposed to [N,K]) ----
  k_f2b<<<ew_grid((size_t)C_*C_),256,0,stream>>>(W_r, bWr, (size_t)C_*C_);
  k_f2b<<<ew_grid((size_t)C_*C_),256,0,stream>>>(W_k, bWk, (size_t)C_*C_);
  k_f2b<<<ew_grid((size_t)C_*C_),256,0,stream>>>(W_v, bWv, (size_t)C_*C_);
  k_f2b<<<ew_grid((size_t)C_*C_),256,0,stream>>>(W_o, bWo, (size_t)C_*C_);
  k_f2b<<<ew_grid((size_t)FF_*C_),256,0,stream>>>(Wkey, bWkey, (size_t)FF_*C_);
  k_f2b<<<ew_grid((size_t)C_*FF_),256,0,stream>>>(Wval, bWval, (size_t)C_*FF_);
  k_tr_f2b<<<ew_grid((size_t)C_*DW_),256,0,stream>>>(w1, bW1, C_, DW_);
  k_tr_f2b<<<ew_grid((size_t)DW_*C_),256,0,stream>>>(w2, bW2, DW_, C_);
  k_tr_f2b<<<ew_grid((size_t)C_*DA_),256,0,stream>>>(a1, bA1, C_, DA_);
  k_tr_f2b<<<ew_grid((size_t)DA_*C_),256,0,stream>>>(a2, bA2, DA_, C_);
  k_tr_f2b<<<ew_grid((size_t)C_*DV_),256,0,stream>>>(v1, bV1, C_, DV_);
  k_tr_f2b<<<ew_grid((size_t)DV_*C_),256,0,stream>>>(v2, bV2, DV_, C_);
  k_tr_f2b<<<ew_grid((size_t)C_*DG_),256,0,stream>>>(g1, bG1, C_, DG_);
  k_tr_f2b<<<ew_grid((size_t)DG_*C_),256,0,stream>>>(g2, bG2, DG_, C_);

  // ---- time-mix front end ----
  k_layernorm<<<M_,256,0,stream>>>(x, ln1w, ln1b, xn, C_);
  k_mix6<<<ew_grid(MC),256,0,stream>>>(xn, x_r, x_w, x_k, x_v, x_a, x_g,
                                       Xr, Xw, Xk, Xv, Xa, Xg);

  const dim3 gemmBlk(128);
  const dim3 gFull(C_/64,  M_/128);   // N=1024
  const dim3 gFFN (FF_/64, M_/128);   // N=4096
  const dim3 gN64 (1,      M_/128);   // N<=64
  const dim3 gN160(3,      M_/128);   // N=160
  k_gemm_bf16<<<gFull,gemmBlk,0,stream>>>(Xr, bWr, rbuf, M_, C_, C_);
  k_gemm_bf16<<<gFull,gemmBlk,0,stream>>>(Xk, bWk, kbuf, M_, C_, C_);
  k_gemm_bf16<<<gFull,gemmBlk,0,stream>>>(Xv, bWv, vraw, M_, C_, C_);

  // w-LoRA: wl2 = tanh(Xw@w1)@w2
  k_gemm_bf16<<<gN64,gemmBlk,0,stream>>>(Xw, bW1, wl1, M_, DW_, C_);
  k_tanh_f2b<<<ew_grid((size_t)M_*DW_),256,0,stream>>>(wl1, wl1b, (size_t)M_*DW_);
  k_gemm_bf16<<<gFull,gemmBlk,0,stream>>>(wl1b, bW2, wl2, M_, C_, DW_);
  // a-LoRA
  k_gemm_bf16<<<gN64,gemmBlk,0,stream>>>(Xa, bA1, al1, M_, DA_, C_);
  k_f2b<<<ew_grid((size_t)M_*DA_),256,0,stream>>>(al1, al1b, (size_t)M_*DA_);
  k_gemm_bf16<<<gFull,gemmBlk,0,stream>>>(al1b, bA2, al2, M_, C_, DA_);
  // v-LoRA
  k_gemm_bf16<<<gN64,gemmBlk,0,stream>>>(Xv, bV1, vl1, M_, DV_, C_);
  k_f2b<<<ew_grid((size_t)M_*DV_),256,0,stream>>>(vl1, vl1b, (size_t)M_*DV_);
  k_gemm_bf16<<<gFull,gemmBlk,0,stream>>>(vl1b, bV2, vl2, M_, C_, DV_);
  // g-LoRA (sigmoid between)
  k_gemm_bf16<<<gN160,gemmBlk,0,stream>>>(Xg, bG1, gl1, M_, DG_, C_);
  k_sigm_f2b<<<ew_grid((size_t)M_*DG_),256,0,stream>>>(gl1, gl1b, (size_t)M_*DG_);
  k_gemm_bf16<<<gFull,gemmBlk,0,stream>>>(gl1b, bG2, gbuf, M_, C_, DG_);

  // ---- WKV7 recurrence ----
  const int total_heads = M_ * H_;   // 65536
  k_prep_scan<<<total_heads/8,256,0,stream>>>(kbuf, vraw, vl2, wl2, al2, vfirst,
                                              k_k, k_a, a0, v0, w0,
                                              decay, knew, vfin, aab, bbb, total_heads);
  k_wkv7_scan<<<B_*H_,256,0,stream>>>(s0, decay, rbuf, knew, vfin, aab, bbb, ybuf);
  k_gn_gate<<<total_heads/8,256,0,stream>>>(ybuf, rbuf, knew, vfin, r_k,
                                            lnxw, lnxb, gbuf, ygb, total_heads);

  // ---- output projection + residual ----
  k_gemm_bf16<<<gFull,gemmBlk,0,stream>>>(ygb, bWo, att, M_, C_, C_);
  k_add<<<ew_grid(MC),256,0,stream>>>(x, att, x1, MC);

  // ---- channel-mix (FFN) ----
  k_layernorm<<<M_,256,0,stream>>>(x1, ln2w, ln2b, xn2, C_);
  k_mix1<<<ew_grid(MC),256,0,stream>>>(xn2, mixk, kfb);
  k_gemm_bf16<<<gFFN,gemmBlk,0,stream>>>(kfb, bWkey, ffn1, M_, FF_, C_);
  k_relu2_f2b<<<ew_grid((size_t)M_*FF_),256,0,stream>>>(ffn1, ffn1b, (size_t)M_*FF_);
  k_gemm_bf16<<<gFull,gemmBlk,0,stream>>>(ffn1b, bWval, ffn2, M_, C_, FF_);

  float* out = (float*)d_out;
  k_add<<<ew_grid(MC),256,0,stream>>>(x1, ffn2, out, MC);
  k_copy<<<ew_grid(MC),256,0,stream>>>(vfirst, out + MC, MC);
}
